// PianoMuseRWKV_27444841021798
// MI455X (gfx1250) — compile-verified, matrix operand order checked by reading, not run on hardware
//
#include <hip/hip_runtime.h>
#include <hip/hip_bf16.h>

// ---------------------------------------------------------------------------
// RWKV-7 forward, MI455X (gfx1250, wave32, WMMA + TDM). Compile-only target.
// GEMMs: bf16 WMMA 16x16x32 fp32-accum; fp32 weights converted to bf16 while
// staging to LDS (HBM-bound: ~750MB fp32 weights -> ~32us floor @ 23.3TB/s).
// A tiles staged via tensor_load_to_lds (TDM) when available; fragments read
// as ds_load_b128 from padded LDS. Scan: thread-per-state-row, LDS broadcast.
// ---------------------------------------------------------------------------

typedef __bf16 bf16_t;
typedef __attribute__((ext_vector_type(16))) __bf16 v16bf;
typedef __attribute__((ext_vector_type(8)))  __bf16 v8bf;
typedef __attribute__((ext_vector_type(8)))  float  v8f;
typedef __attribute__((ext_vector_type(4)))  float  v4f;
typedef __attribute__((ext_vector_type(4)))  unsigned int v4u;
typedef __attribute__((ext_vector_type(8)))  int    v8i;
typedef __attribute__((ext_vector_type(4)))  int    v4i;

#define TT 512
#define CCH 768
#define HH 12
#define NH 64
#define VV 65536
#define CT (TT * CCH)

#if defined(__gfx1250__) && __has_builtin(__builtin_amdgcn_tensor_load_to_lds)
#define TDM_OK 1
#else
#define TDM_OK 0
#endif

enum { EP_NONE = 0, EP_TANH = 1, EP_SIGMOID = 2, EP_RELUSQ = 3 };

__device__ __forceinline__ float sigmoidf_(float x) {
  return 1.f / (1.f + __expf(-x));
}

// ----------------------------- LayerNorm -----------------------------------
template <typename OutT>
__global__ void rwkv_ln(const float* __restrict__ x, const float* __restrict__ w,
                        const float* __restrict__ b, OutT* __restrict__ y) {
  int token = blockIdx.x * 8 + (threadIdx.x >> 5);
  int lane = threadIdx.x & 31;
  if (token >= TT) return;
  const float* xr = x + (long)token * CCH;
  float s = 0.f, ss = 0.f;
  for (int c = lane; c < CCH; c += 32) {
    float v = xr[c];
    s += v;
    ss += v * v;
  }
#pragma unroll
  for (int m = 16; m > 0; m >>= 1) {
    s += __shfl_xor(s, m, 32);
    ss += __shfl_xor(ss, m, 32);
  }
  float mean = s * (1.f / CCH);
  float var = ss * (1.f / CCH) - mean * mean;
  float inv = rsqrtf(var + 1e-5f);
  OutT* yr = y + (long)token * CCH;
  for (int c = lane; c < CCH; c += 32) {
    yr[c] = (OutT)((xr[c] - mean) * inv * w[c] + b[c]);
  }
}

// ----------------------------- Embedding -----------------------------------
__global__ void rwkv_embed(const int* __restrict__ ids,
                           const float* __restrict__ emb,
                           float* __restrict__ x, float* __restrict__ xprev) {
  int t = blockIdx.x;
  long row = (long)ids[t] * CCH;
  long rowp = t ? (long)ids[t - 1] * CCH : 0;
  for (int c = threadIdx.x; c < CCH; c += blockDim.x) {
    x[(long)t * CCH + c] = emb[row + c];
    xprev[(long)t * CCH + c] = t ? emb[rowp + c] : 0.f;
  }
}

// ----------------------------- 6-way token mix ------------------------------
__global__ void rwkv_mix6(const float* __restrict__ xx, const float* __restrict__ xxp,
                          const float* __restrict__ pr, const float* __restrict__ pw,
                          const float* __restrict__ pk, const float* __restrict__ pv,
                          const float* __restrict__ pa, const float* __restrict__ pg,
                          bf16_t* __restrict__ mr, bf16_t* __restrict__ mw,
                          bf16_t* __restrict__ mk, bf16_t* __restrict__ mv,
                          bf16_t* __restrict__ ma, bf16_t* __restrict__ mg) {
  int i = blockIdx.x * blockDim.x + threadIdx.x;
  if (i >= CT) return;
  int c = i % CCH;
  float a = xx[i], d = xxp[i] - a;
  mr[i] = (bf16_t)(a + d * pr[c]);
  mw[i] = (bf16_t)(a + d * pw[c]);
  mk[i] = (bf16_t)(a + d * pk[c]);
  mv[i] = (bf16_t)(a + d * pv[c]);
  ma[i] = (bf16_t)(a + d * pa[c]);
  mg[i] = (bf16_t)(a + d * pg[c]);
}

// ----------------------------- WMMA GEMM -----------------------------------
// C[M,N] = epilogue(A[M,K](bf16) @ B[K,N](fp32->bf16)) (+bias, +residual)
// BT: B stored [N,K] row-major (ldb=K), for the vocab head.
// Requires: M%64==0, K%32==0, N%8==0 (N=32 handled by zero-fill + col guard).
// Block: 256 threads = 8 waves; block tile 64x64; wave tile 16x32 (2 wmma acc).
template <int EP, typename OutT, bool BT>
__global__ __launch_bounds__(256) void rwkv_gemm(
    const bf16_t* __restrict__ A, const float* __restrict__ Bw,
    OutT* __restrict__ Cout, const float* __restrict__ bias,
    const float* __restrict__ residual, int M, int N, int K, int ldb) {
  // 80B row pitch (32 elems + 16B pad) keeps 16B-aligned b128 LDS accesses.
  __shared__ bf16_t As[64][40];   // A tile, [m][k]
  __shared__ bf16_t BsT[64][40];  // B tile transposed, [n][k]

  int tid = threadIdx.x;
  int lane = tid & 31;
  int waveId = __builtin_amdgcn_readfirstlane(tid >> 5);
  int wm = waveId >> 1;  // 0..3
  int wn = waveId & 1;   // 0..1
  int bM = blockIdx.y * 64;
  int bN = blockIdx.x * 64;

  v8f c0 = {};
  v8f c1 = {};

  for (int k0 = 0; k0 < K; k0 += 32) {
    // ---------------- stage A tile 64x32 (bf16 raw copy) ----------------
#if TDM_OK
    if (waveId == 0) {
      unsigned long long ga =
          (unsigned long long)(const void*)(A + (long)bM * K + k0);
      unsigned ldsa = (unsigned)(size_t)(&As[0][0]);
      v4u g0;
      g0[0] = 1u;                                    // count=1 (user D#)
      g0[1] = ldsa;                                  // lds_addr (bytes)
      g0[2] = (unsigned)(ga & 0xFFFFFFFFu);          // global_addr lo
      g0[3] = (unsigned)((ga >> 32) & 0x01FFFFFFu) | (2u << 30);  // type=2
      v8i g1;
      // data_size=1 (2B), pad_enable, pad_interval=3 (16 DW), pad_amount=3 (4 DW)
      g1[0] = (1 << 16) | (1 << 20) | (3 << 22) | (3 << 25);
      g1[1] = (K & 0xFFFF) << 16;                    // tensor_dim0 lo16
      g1[2] = ((K >> 16) & 0xFFFF) | ((M & 0xFFFF) << 16);  // dim0 hi | dim1 lo
      g1[3] = ((M >> 16) & 0xFFFF) | (32 << 16);     // dim1 hi | tile_dim0=32
      g1[4] = 64;                                    // tile_dim1=64, tile_dim2=0
      g1[5] = K;                                     // tensor_dim0_stride lo32
      g1[6] = 0;
      g1[7] = 0;
      v4i z4 = {0, 0, 0, 0};
#if __clang_major__ >= 23
      v8i z8 = {0, 0, 0, 0, 0, 0, 0, 0};
      __builtin_amdgcn_tensor_load_to_lds(g0, g1, z4, z4, z8, 0);
#else
      __builtin_amdgcn_tensor_load_to_lds(g0, g1, z4, z4, 0);
#endif
      __builtin_amdgcn_s_wait_tensorcnt(0);
    }
#else
    {
      int r = tid >> 2;                 // 0..63 (m)
      int kc = (tid & 3) * 8;           // 0,8,16,24
      *(v8bf*)&As[r][kc] =
          *(const v8bf*)(A + (long)(bM + r) * K + (k0 + kc));
    }
#endif
    // ------- stage B tile (fp32 -> bf16 in regs), transposed in LDS -------
    if (!BT) {
      int r = tid >> 3;                 // 0..31 (k)
      int nc = (tid & 7) * 8;           // 0..56 (n)
      const float* src = Bw + (long)(k0 + r) * ldb + bN + nc;
      v8bf pack;
      if (bN + nc + 8 <= N) {
        v4f f0 = *(const v4f*)(src);
        v4f f1 = *(const v4f*)(src + 4);
#pragma unroll
        for (int j = 0; j < 4; ++j) {
          pack[j] = (bf16_t)f0[j];
          pack[4 + j] = (bf16_t)f1[j];
        }
        if (k0 + 32 < K) __builtin_prefetch(src + 32 * (long)ldb, 0, 1);
      } else {
#pragma unroll
        for (int j = 0; j < 8; ++j) pack[j] = (bf16_t)0.f;
      }
#pragma unroll
      for (int j = 0; j < 8; ++j) BsT[nc + j][r] = pack[j];  // [n][k] scatter
    } else {
      int n = tid >> 2;                 // 0..63
      int kc = (tid & 3) * 8;           // 0..24
      const float* src = Bw + (long)(bN + n) * ldb + k0 + kc;
      v4f f0 = *(const v4f*)(src);
      v4f f1 = *(const v4f*)(src + 4);
      v8bf pack;
#pragma unroll
      for (int j = 0; j < 4; ++j) {
        pack[j] = (bf16_t)f0[j];
        pack[4 + j] = (bf16_t)f1[j];
      }
      *(v8bf*)&BsT[n][kc] = pack;
      if (k0 + 32 < K) __builtin_prefetch(src + 32, 0, 1);
    }
    __syncthreads();

    // ---- fragments (ISA 7.12.2): lane<16 K in {0..7,16..23}, else +8 ----
    int mrow = wm * 16 + (lane & 15);
    int kb = (lane >> 4) * 8;  // 0 or 8
    v8bf alo = *(const v8bf*)&As[mrow][kb];
    v8bf ahi = *(const v8bf*)&As[mrow][kb + 16];
    v16bf af = __builtin_shufflevector(alo, ahi, 0, 1, 2, 3, 4, 5, 6, 7, 8, 9,
                                       10, 11, 12, 13, 14, 15);
    int nc0 = wn * 32 + (lane & 15);
    v8bf b0lo = *(const v8bf*)&BsT[nc0][kb];
    v8bf b0hi = *(const v8bf*)&BsT[nc0][kb + 16];
    v16bf b0f = __builtin_shufflevector(b0lo, b0hi, 0, 1, 2, 3, 4, 5, 6, 7, 8,
                                        9, 10, 11, 12, 13, 14, 15);
    v8bf b1lo = *(const v8bf*)&BsT[nc0 + 16][kb];
    v8bf b1hi = *(const v8bf*)&BsT[nc0 + 16][kb + 16];
    v16bf b1f = __builtin_shufflevector(b1lo, b1hi, 0, 1, 2, 3, 4, 5, 6, 7, 8,
                                        9, 10, 11, 12, 13, 14, 15);

    c0 = __builtin_amdgcn_wmma_f32_16x16x32_bf16(false, af, false, b0f,
                                                 (short)0, c0, false, false);
    c1 = __builtin_amdgcn_wmma_f32_16x16x32_bf16(false, af, false, b1f,
                                                 (short)0, c1, false, false);
    __syncthreads();
  }

  // ---- epilogue: C/D layout: lane half -> M+8; elem i -> row + i ----
  int rb = bM + wm * 16 + ((lane >> 4) * 8);
  int cb = bN + wn * 32 + (lane & 15);
#pragma unroll
  for (int i = 0; i < 8; ++i) {
    int row = rb + i;
    {
      int col = cb;
      if (col < N) {
        float v = c0[i];
        if (bias) v += bias[col];
        if (EP == EP_TANH) v = tanhf(v);
        else if (EP == EP_SIGMOID) v = sigmoidf_(v);
        else if (EP == EP_RELUSQ) { float t = fmaxf(v, 0.f); v = t * t; }
        if (residual) v += residual[(long)row * N + col];
        Cout[(long)row * N + col] = (OutT)v;
      }
    }
    {
      int col = cb + 16;
      if (col < N) {
        float v = c1[i];
        if (bias) v += bias[col];
        if (EP == EP_TANH) v = tanhf(v);
        else if (EP == EP_SIGMOID) v = sigmoidf_(v);
        else if (EP == EP_RELUSQ) { float t = fmaxf(v, 0.f); v = t * t; }
        if (residual) v += residual[(long)row * N + col];
        Cout[(long)row * N + col] = (OutT)v;
      }
    }
  }
}

// ----------------------------- WKV prep ------------------------------------
__global__ void rwkv_wkv_prep(const float* __restrict__ kraw,
                              const float* __restrict__ a_s,
                              const float* __restrict__ wlo,
                              const float* __restrict__ vraw,
                              const float* __restrict__ vg_s,
                              const float* __restrict__ w0,
                              const float* __restrict__ k_k,
                              const float* __restrict__ k_a,
                              float* __restrict__ wd, float* __restrict__ kk,
                              float* __restrict__ kmod, float* __restrict__ vfin,
                              float* __restrict__ vfirst, int is_first) {
  int t = blockIdx.x / HH;
  int h = blockIdx.x % HH;
  int j = threadIdx.x;  // 0..63
  int c = h * NH + j;
  long idx = (long)t * CCH + c;

  float k = kraw[idx];
  float kk0 = k * k_k[c];

  __shared__ float red[64];
  red[j] = kk0 * kk0;
  __syncthreads();
#pragma unroll
  for (int s = 32; s > 0; s >>= 1) {
    if (j < s) red[j] += red[j + s];
    __syncthreads();
  }
  float nrm = sqrtf(red[0]);
  kk[idx] = kk0 / fmaxf(nrm, 1e-12f);

  float a = a_s[idx];
  kmod[idx] = k * (1.f + (a - 1.f) * k_a[c]);
  // exp(-softplus(-(w0+wlo)) - 0.5) == e^{-0.5} * sigmoid(w0+wlo)
  wd[idx] = 0.60653065971263342f * sigmoidf_(w0[c] + wlo[idx]);

  float v = vraw[idx];
  if (is_first) {
    vfirst[idx] = v;
    vfin[idx] = v;
  } else {
    float s = vg_s[idx];
    vfin[idx] = v + (vfirst[idx] - v) * s;
  }
}

// ----------------------------- WKV scan ------------------------------------
// One block per head, 64 threads; thread i owns state row i (64 fp32 regs).
__global__ void rwkv_wkv_scan(const float* __restrict__ r,
                              const float* __restrict__ wdp,
                              const float* __restrict__ kp,
                              const float* __restrict__ vp,
                              const float* __restrict__ kkp,
                              const float* __restrict__ ap,
                              float* __restrict__ out) {
  int h = blockIdx.x;
  int i = threadIdx.x;  // 0..63
  float s[64];
#pragma unroll
  for (int j = 0; j < 64; ++j) s[j] = 0.f;

  __shared__ float rs[64], ws[64], ks[64], kks[64], as[64];
  int base = h * NH;
  for (int t = 0; t < TT; ++t) {
    long idx = (long)t * CCH + base + i;
    __syncthreads();
    rs[i] = r[idx];
    ws[i] = wdp[idx];
    ks[i] = kp[idx];
    kks[i] = kkp[idx];
    as[i] = ap[idx];
    float vi = vp[idx];
    __syncthreads();

    float sa = 0.f;
#pragma unroll
    for (int j = 0; j < 64; ++j) sa += s[j] * kks[j];
    float o = 0.f;
#pragma unroll
    for (int j = 0; j < 64; ++j) {
      s[j] = s[j] * ws[j] - sa * (kks[j] * as[j]) + vi * ks[j];
      o += s[j] * rs[j];
    }
    out[idx] = o;
  }
}

// ----------------------------- post GroupNorm + rk --------------------------
__global__ void rwkv_postgn(const float* __restrict__ outscan,
                            const float* __restrict__ r,
                            const float* __restrict__ kmod,
                            const float* __restrict__ vfin,
                            const float* __restrict__ g,
                            const float* __restrict__ lnxw,
                            const float* __restrict__ lnxb,
                            const float* __restrict__ r_k,
                            bf16_t* __restrict__ og) {
  int t = blockIdx.x / HH;
  int h = blockIdx.x % HH;
  int j = threadIdx.x;
  int c = h * NH + j;
  long idx = (long)t * CCH + c;

  float o = outscan[idx];
  float rk = r[idx] * kmod[idx] * r_k[c];

  __shared__ float red[64];
  red[j] = o;
  __syncthreads();
#pragma unroll
  for (int s = 32; s > 0; s >>= 1) {
    if (j < s) red[j] += red[j + s];
    __syncthreads();
  }
  float mean = red[0] * (1.f / NH);
  __syncthreads();
  float d = o - mean;
  red[j] = d * d;
  __syncthreads();
#pragma unroll
  for (int s = 32; s > 0; s >>= 1) {
    if (j < s) red[j] += red[j + s];
    __syncthreads();
  }
  float var = red[0] * (1.f / NH);
  __syncthreads();
  red[j] = rk;
  __syncthreads();
#pragma unroll
  for (int s = 32; s > 0; s >>= 1) {
    if (j < s) red[j] += red[j + s];
    __syncthreads();
  }
  float rks = red[0];

  float gn = d * rsqrtf(var + 1e-5f) * lnxw[c] + lnxb[c];
  float res = gn + rks * vfin[idx];
  og[idx] = (bf16_t)(res * g[idx]);
}

// ----------------------------- channel-mix input ----------------------------
__global__ void rwkv_cmix(const float* __restrict__ xx2,
                          const float* __restrict__ fxk,
                          bf16_t* __restrict__ outb) {
  int i = blockIdx.x * blockDim.x + threadIdx.x;
  if (i >= CT) return;
  int t = i / CCH, c = i % CCH;
  float cur = xx2[i];
  float prev = t ? xx2[i - CCH] : 0.f;
  outb[i] = (bf16_t)(cur + (prev - cur) * fxk[c]);
}

// ----------------------------- host-side helpers ----------------------------
static inline void g_f32(hipStream_t s, const bf16_t* A, const float* B, float* C,
                         int M, int N, int K, int ep, const float* bias,
                         const float* res) {
  dim3 gr((N + 63) / 64, (M + 63) / 64);
  switch (ep) {
    case EP_SIGMOID:
      rwkv_gemm<EP_SIGMOID, float, false><<<gr, 256, 0, s>>>(A, B, C, bias, res, M, N, K, N);
      break;
    default:
      rwkv_gemm<EP_NONE, float, false><<<gr, 256, 0, s>>>(A, B, C, bias, res, M, N, K, N);
      break;
  }
}

static inline void g_bf16(hipStream_t s, const bf16_t* A, const float* B, bf16_t* C,
                          int M, int N, int K, int ep) {
  dim3 gr((N + 63) / 64, (M + 63) / 64);
  switch (ep) {
    case EP_TANH:
      rwkv_gemm<EP_TANH, bf16_t, false><<<gr, 256, 0, s>>>(A, B, C, nullptr, nullptr, M, N, K, N);
      break;
    case EP_SIGMOID:
      rwkv_gemm<EP_SIGMOID, bf16_t, false><<<gr, 256, 0, s>>>(A, B, C, nullptr, nullptr, M, N, K, N);
      break;
    case EP_RELUSQ:
      rwkv_gemm<EP_RELUSQ, bf16_t, false><<<gr, 256, 0, s>>>(A, B, C, nullptr, nullptr, M, N, K, N);
      break;
    default:
      rwkv_gemm<EP_NONE, bf16_t, false><<<gr, 256, 0, s>>>(A, B, C, nullptr, nullptr, M, N, K, N);
      break;
  }
}

struct BlkP {
  const float *ln1_w, *ln1_b, *ln2_w, *ln2_b;
  const float *x_r, *x_w, *x_k, *x_v, *x_a, *x_g;
  const float *Wr, *Wk, *Wv, *Wo;
  const float *w0, *w1, *w2, *a0, *a1, *a2, *g1, *g2;
  const float *k_k, *k_a, *r_k, *ln_x_w, *ln_x_b;
  const float *ffn_x_k, *ffn_Wk, *ffn_Wv;
  const float *v0, *v1, *v2;
};

extern "C" void kernel_launch(void* const* d_in, const int* in_sizes, int n_in,
                              void* d_out, int out_size, void* d_ws, size_t ws_size,
                              hipStream_t stream) {
  (void)in_sizes; (void)n_in; (void)out_size; (void)ws_size;
  int idx = 0;
  const int* ids = (const int*)d_in[idx++];
  const float* emb = (const float*)d_in[idx++];
  const float* head = (const float*)d_in[idx++];
  const float* ln_out_w = (const float*)d_in[idx++];
  const float* ln_out_b = (const float*)d_in[idx++];

  BlkP blk[12];
  for (int i = 0; i < 12; ++i) {
    BlkP& b = blk[i];
    b.ln1_w = (const float*)d_in[idx++];  b.ln1_b = (const float*)d_in[idx++];
    b.ln2_w = (const float*)d_in[idx++];  b.ln2_b = (const float*)d_in[idx++];
    b.x_r = (const float*)d_in[idx++];    b.x_w = (const float*)d_in[idx++];
    b.x_k = (const float*)d_in[idx++];    b.x_v = (const float*)d_in[idx++];
    b.x_a = (const float*)d_in[idx++];    b.x_g = (const float*)d_in[idx++];
    b.Wr = (const float*)d_in[idx++];     b.Wk = (const float*)d_in[idx++];
    b.Wv = (const float*)d_in[idx++];     b.Wo = (const float*)d_in[idx++];
    b.w0 = (const float*)d_in[idx++];     b.w1 = (const float*)d_in[idx++];
    b.w2 = (const float*)d_in[idx++];
    b.a0 = (const float*)d_in[idx++];     b.a1 = (const float*)d_in[idx++];
    b.a2 = (const float*)d_in[idx++];
    b.g1 = (const float*)d_in[idx++];     b.g2 = (const float*)d_in[idx++];
    b.k_k = (const float*)d_in[idx++];    b.k_a = (const float*)d_in[idx++];
    b.r_k = (const float*)d_in[idx++];
    b.ln_x_w = (const float*)d_in[idx++]; b.ln_x_b = (const float*)d_in[idx++];
    b.ffn_x_k = (const float*)d_in[idx++];
    b.ffn_Wk = (const float*)d_in[idx++]; b.ffn_Wv = (const float*)d_in[idx++];
    if (i > 0) {
      b.v0 = (const float*)d_in[idx++];
      b.v1 = (const float*)d_in[idx++];
      b.v2 = (const float*)d_in[idx++];
    } else {
      b.v0 = b.v1 = b.v2 = nullptr;
    }
  }

  // ---- workspace carve (256B aligned) ----
  char* wp = (char*)d_ws;
  auto alloc = [&](size_t bytes) -> void* {
    void* p = (void*)wp;
    wp += (bytes + 255) & ~(size_t)255;
    return p;
  };
  float* x      = (float*)alloc(CT * 4);
  float* xprev  = (float*)alloc(CT * 4);
  float* xxA    = (float*)alloc(CT * 4);
  float* xxB    = (float*)alloc(CT * 4);
  float* rbuf   = (float*)alloc(CT * 4);
  float* kraw   = (float*)alloc(CT * 4);
  float* vraw   = (float*)alloc(CT * 4);
  float* a_s    = (float*)alloc(CT * 4);
  float* gbuf   = (float*)alloc(CT * 4);
  float* wlo    = (float*)alloc(CT * 4);
  float* vg     = (float*)alloc(CT * 4);
  float* wd     = (float*)alloc(CT * 4);
  float* kkb    = (float*)alloc(CT * 4);
  float* kmod   = (float*)alloc(CT * 4);
  float* vfin   = (float*)alloc(CT * 4);
  float* vfirst = (float*)alloc(CT * 4);
  float* oscan  = (float*)alloc(CT * 4);
  float* xx2    = (float*)alloc(CT * 4);
  bf16_t* mr    = (bf16_t*)alloc(CT * 2);
  bf16_t* mw    = (bf16_t*)alloc(CT * 2);
  bf16_t* mk    = (bf16_t*)alloc(CT * 2);
  bf16_t* mv    = (bf16_t*)alloc(CT * 2);
  bf16_t* ma    = (bf16_t*)alloc(CT * 2);
  bf16_t* mg    = (bf16_t*)alloc(CT * 2);
  bf16_t* og    = (bf16_t*)alloc(CT * 2);
  bf16_t* cmixb = (bf16_t*)alloc(CT * 2);
  bf16_t* xlnb  = (bf16_t*)alloc(CT * 2);
  bf16_t* h64a  = (bf16_t*)alloc((size_t)TT * 64 * 2);
  bf16_t* h64b  = (bf16_t*)alloc((size_t)TT * 64 * 2);
  bf16_t* h32   = (bf16_t*)alloc((size_t)TT * 32 * 2);
  bf16_t* h128  = (bf16_t*)alloc((size_t)TT * 128 * 2);
  bf16_t* ffnh  = (bf16_t*)alloc((size_t)TT * 3072 * 2);

  // ---- pipeline ----
  rwkv_embed<<<TT, 256, 0, stream>>>(ids, emb, x, xprev);

  for (int i = 0; i < 12; ++i) {
    const BlkP& b = blk[i];
    rwkv_ln<float><<<TT / 8, 256, 0, stream>>>(x, b.ln1_w, b.ln1_b, xxA);
    rwkv_ln<float><<<TT / 8, 256, 0, stream>>>(xprev, b.ln1_w, b.ln1_b, xxB);
    rwkv_mix6<<<(CT + 255) / 256, 256, 0, stream>>>(xxA, xxB, b.x_r, b.x_w, b.x_k,
                                                    b.x_v, b.x_a, b.x_g, mr, mw, mk,
                                                    mv, ma, mg);
    // big projections
    g_f32(stream, mr, b.Wr, rbuf, TT, CCH, CCH, EP_NONE, nullptr, nullptr);
    g_f32(stream, mk, b.Wk, kraw, TT, CCH, CCH, EP_NONE, nullptr, nullptr);
    g_f32(stream, mv, b.Wv, vraw, TT, CCH, CCH, EP_NONE, nullptr, nullptr);
    // decay path: w_lo = tanh(mw@w1)@w2
    g_bf16(stream, mw, b.w1, h64a, TT, 64, CCH, EP_TANH);
    g_f32(stream, h64a, b.w2, wlo, TT, CCH, 64, EP_NONE, nullptr, nullptr);
    // a = sigmoid(a0 + (ma@a1)@a2)
    g_bf16(stream, ma, b.a1, h64b, TT, 64, CCH, EP_NONE);
    g_f32(stream, h64b, b.a2, a_s, TT, CCH, 64, EP_SIGMOID, b.a0, nullptr);
    // g = sigmoid(mg@g1)@g2
    g_bf16(stream, mg, b.g1, h128, TT, 128, CCH, EP_SIGMOID);
    g_f32(stream, h128, b.g2, gbuf, TT, CCH, 128, EP_NONE, nullptr, nullptr);
    // v gate (layers > 0)
    if (i > 0) {
      g_bf16(stream, mv, b.v1, h32, TT, 32, CCH, EP_NONE);
      g_f32(stream, h32, b.v2, vg, TT, CCH, 32, EP_SIGMOID, b.v0, nullptr);
    }
    rwkv_wkv_prep<<<TT * HH, 64, 0, stream>>>(kraw, a_s, wlo, vraw, vg, b.w0, b.k_k,
                                              b.k_a, wd, kkb, kmod, vfin, vfirst,
                                              i == 0 ? 1 : 0);
    rwkv_wkv_scan<<<HH, 64, 0, stream>>>(rbuf, wd, kmod, vfin, kkb, a_s, oscan);
    rwkv_postgn<<<TT * HH, 64, 0, stream>>>(oscan, rbuf, kmod, vfin, gbuf, b.ln_x_w,
                                            b.ln_x_b, b.r_k, og);
    // x += (out*g) @ Wo
    g_f32(stream, og, b.Wo, x, TT, CCH, CCH, EP_NONE, nullptr, x);
    // x_prev = x (post time-mix residual)
    hipMemcpyAsync(xprev, x, (size_t)CT * 4, hipMemcpyDeviceToDevice, stream);
    // channel mix
    rwkv_ln<float><<<TT / 8, 256, 0, stream>>>(x, b.ln2_w, b.ln2_b, xx2);
    rwkv_cmix<<<(CT + 255) / 256, 256, 0, stream>>>(xx2, b.ffn_x_k, cmixb);
    g_bf16(stream, cmixb, b.ffn_Wk, ffnh, TT, 4 * CCH, CCH, EP_RELUSQ);
    g_f32(stream, ffnh, b.ffn_Wv, x, TT, CCH, 4 * CCH, EP_NONE, nullptr, x);
  }

  rwkv_ln<bf16_t><<<TT / 8, 256, 0, stream>>>(x, ln_out_w, ln_out_b, xlnb);
  // head: out[t,v] = sum_c xln[t,c] * head[v,c]  (B transposed, ldb = C)
  {
    dim3 gr(VV / 64, TT / 64);
    rwkv_gemm<EP_NONE, float, true><<<gr, 256, 0, stream>>>(
        xlnb, head, (float*)d_out, nullptr, nullptr, TT, VV, CCH, CCH);
  }
}
